// CoSADUV_51007031608049
// MI455X (gfx1250) — matrix operand order repaired and checked by reading, not compiled, not run on hardware
//
#include <hip/hip_runtime.h>
#include <hip/hip_bf16.h>
#include <math.h>

// ---------------------------------------------------------------------------
// CoSADUV ReNet saliency pipeline for MI455X (gfx1250, wave32, WMMA).
// - GEMMs use v_wmma_f32_16x16x32_bf16 (bf16 operands, f32 accum).
// - LDS operand tiles are stored in WMMA fragment order: one contiguous,
//   32B-aligned v16bf per lane (2x ds_load_b128 per fragment).
// - Staging is two-phase (float4 loads -> v4bf ds_store_b64) so loads clause.
// - Projection GEMM M is t-major (m = t*RPAD + r): each 64-tile sits in one
//   timestep, pre is stored [t][gate][r] -> scan's per-step pre fetch is an
//   aligned v8f load per n-tile matching the WMMA C-fragment layout.
// - The scan software-pipelines pre[t+1] loads behind WMMA + gates.
// - Whh^T lives in LDS (128KB of the 320KB WGP LDS).
// - The 368MB temporal GEMV is HBM-bound (~16us @ 23.3TB/s) -> float4 FMA.
// ---------------------------------------------------------------------------

typedef __bf16 bf16_t;
typedef __attribute__((ext_vector_type(16))) __bf16 v16bf;
typedef __attribute__((ext_vector_type(4)))  __bf16 v4bf;
typedef __attribute__((ext_vector_type(8)))  float  v8f;

__device__ __forceinline__ float sigmf(float x) { return 1.0f / (1.0f + __expf(-x)); }

__device__ __forceinline__ v16bf load_frag(const bf16_t* p) {
  return *(const v16bf*)p;
}

// A-tile swizzle (16x32 MxK bf16): lane L (m=L&15, half=L>>4)
// owns local k = half*16 + {0..15}.  4 consecutive k
// (aligned to 4) stay contiguous.
__device__ __forceinline__ int swizA(int mtile, int m, int k) {
  const int half = k >> 4;
  const int pos  = k & 15;
  return ((mtile * 2 + half) * 16 + m) * 16 + pos;
}

// B-tile swizzle (32x16 KxN bf16): lane L (n=L&15, half=L>>4) owns
// k = half*16 + {0..15}.  4 consecutive k (aligned) stay contiguous.
__device__ __forceinline__ int swizB(int ntile, int n, int k) {
  const int half = k >> 4;
  return ((ntile * 2 + half) * 16 + n) * 16 + (k & 15);
}

__device__ __forceinline__ v4bf cvt4(float4 f) {
  v4bf b;
  b[0] = (bf16_t)f.x; b[1] = (bf16_t)f.y; b[2] = (bf16_t)f.z; b[3] = (bf16_t)f.w;
  return b;
}

__device__ __forceinline__ v8f wmma_bf16(v16bf a, v16bf b, v8f c) {
  return __builtin_amdgcn_wmma_f32_16x16x32_bf16(false, a, false, b, (short)0, c,
                                                 false, false);
}

// ---------------------------------------------------------------------------
// ctx1 = scene @ fc1_W^T + fc1_b  [2,512];  ctxR = scene @ fcR_W^T + fcR_b [2,256]
// ---------------------------------------------------------------------------
__global__ void ctx_kernel(const float* __restrict__ scene,
                           const float* __restrict__ fc1W, const float* __restrict__ fc1b,
                           const float* __restrict__ fcRW, const float* __restrict__ fcRb,
                           float* __restrict__ ctx1, float* __restrict__ ctxR) {
  const int idx = blockIdx.x * blockDim.x + threadIdx.x;
  if (idx < 1024) {
    const int n = idx >> 9, j = idx & 511;
    float s = fc1b[j];
    for (int k = 0; k < 128; ++k) s += scene[n * 128 + k] * fc1W[j * 128 + k];
    ctx1[idx] = s;
  } else if (idx < 1536) {
    const int t = idx - 1024;
    const int n = t >> 8, j = t & 255;
    float s = fcRb[j];
    for (int k = 0; k < 128; ++k) s += scene[n * 128 + k] * fcRW[j * 128 + k];
    ctxR[n * 256 + j] = s;
  }
}

// ---------------------------------------------------------------------------
// Input projection GEMM, t-major M:  pre[t][gate][r] = seq(r,t,:)@W^T + b.
// R = 2*RPN real rows padded to RPAD (multiple of 64) -> each 64-tile is
// inside one timestep; ctx token at t==0 / t==T-1 (block-uniform).
// 64x64 block tile, 4 waves x (16M x 64N), bf16 staged swizzled through LDS.
// CONTIG: channel stride == 1 (NHWC passes) -> float4 A loads.
// ---------------------------------------------------------------------------
template <int T, int I, int RPN, int RPAD, bool CONTIG>
__global__ void __launch_bounds__(128)
proj_kernel(const float* __restrict__ X, const float* __restrict__ ctx,
            const float* __restrict__ Wf, const float* __restrict__ Wb,
            const float* __restrict__ biasf, const float* __restrict__ biasb,
            float* __restrict__ preF, float* __restrict__ preB,
            long sn, long sr, long st, long sc) {
  __shared__ __align__(32) bf16_t As[64 * 32];  // swizA layout, mtile=0..3
  __shared__ __align__(32) bf16_t Bs[32 * 64];  // swizB layout, ntile=0..3
  constexpr int R = 2 * RPN;
  const int    dir  = blockIdx.z;
  const float* W    = dir ? Wb : Wf;
  const float* bias = dir ? biasb : biasf;
  float*       pre  = dir ? preB : preF;
  const int m0      = blockIdx.x * 64;
  const int n0      = blockIdx.y * 64;
  const int wave    = threadIdx.x >> 5;
  const int lane_hi = (threadIdx.x >> 4) & 1;
  const int lane_lo = threadIdx.x & 15;

  const int  tb    = m0 / RPAD;           // block-uniform timestep
  const int  rb    = m0 - tb * RPAD;      // first row of this tile
  const bool isctx = (tb == 0) || (tb == T - 1);

  v8f acc[4] = {};

  for (int k0 = 0; k0 < I; k0 += 32) {
    // ---- A tile (64 rows x 32 k): phase 1 = batched loads, phase 2 = stores.
    float4 abuf[4];
#pragma unroll
    for (int q = 0; q < 4; ++q) {
      const int eq = q * 128 + (int)threadIdx.x;   // 512 quads
      const int mi = eq >> 3, kq = eq & 7;
      const int r  = rb + mi;
      const int rc = (r < R) ? r : (R - 1);
      const int c  = k0 + kq * 4;
      if (isctx) {
        abuf[q] = *(const float4*)(ctx + (rc / RPN) * I + c);
      } else if (CONTIG) {
        abuf[q] = *(const float4*)(X + (long)(rc / RPN) * sn +
                                   (long)(rc % RPN) * sr + (long)(tb - 1) * st + c);
      } else {
        const float* p = X + (long)(rc / RPN) * sn + (long)(rc % RPN) * sr +
                         (long)(tb - 1) * st + (long)c * sc;
        abuf[q] = make_float4(p[0], p[sc], p[2 * sc], p[3 * sc]);
      }
    }
#pragma unroll
    for (int q = 0; q < 4; ++q) {
      const int eq = q * 128 + (int)threadIdx.x;
      const int mi = eq >> 3, kq = eq & 7;
      float4 f = abuf[q];
      if (rb + mi >= R) f = make_float4(0.f, 0.f, 0.f, 0.f);
      *(v4bf*)(As + swizA(mi >> 4, mi & 15, kq * 4)) = cvt4(f);
    }
    // ---- B tile (32 k x 64 n), k-fastest reads (contiguous in W).
    float4 bbuf[4];
#pragma unroll
    for (int q = 0; q < 4; ++q) {
      const int eq = q * 128 + (int)threadIdx.x;
      const int ni = eq >> 3, kq = eq & 7;
      bbuf[q] = *(const float4*)(W + (long)(n0 + ni) * I + (k0 + kq * 4));
    }
#pragma unroll
    for (int q = 0; q < 4; ++q) {
      const int eq = q * 128 + (int)threadIdx.x;
      const int ni = eq >> 3, kq = eq & 7;
      *(v4bf*)(Bs + swizB(ni >> 4, ni & 15, kq * 4)) = cvt4(bbuf[q]);
    }
    __syncthreads();

    const v16bf af = load_frag(As + ((wave * 2 + lane_hi) * 16 + lane_lo) * 16);
#pragma unroll
    for (int nt = 0; nt < 4; ++nt) {
      const v16bf bfg = load_frag(Bs + ((nt * 2 + lane_hi) * 16 + lane_lo) * 16);
      acc[nt] = wmma_bf16(af, bfg, acc[nt]);
    }
    __syncthreads();
  }

  // Store pre[t][n][r]; C-frag element v <-> row rb + wave*16 + v + 8*lane_hi.
#pragma unroll
  for (int nt = 0; nt < 4; ++nt) {
    const int n = n0 + nt * 16 + lane_lo;
#pragma unroll
    for (int v = 0; v < 8; ++v) {
      const int r = rb + wave * 16 + v + 8 * lane_hi;
      pre[((long)tb * 512 + n) * RPAD + r] = acc[nt][v] + bias[n];
    }
  }
}

// ---------------------------------------------------------------------------
// LSTM scan: 16 independent sequences per workgroup, T serial steps.
// Per step: g[16,512] = pre + h_prev[16,128] @ Whh^T (WMMA bf16, Whh^T kept
// in LDS in fragment order). pre[t][n][r] layout -> C-fragment init is a
// single aligned v8f load per n-tile; pre[t+1] is prefetched into registers
// while the current step's WMMA + gates run.
// LDS: WhhB 128KB + hs 4KB + gmat 32KB + c-state 8KB = 172KB (< 320KB/WGP).
// ---------------------------------------------------------------------------
template <int T, int RPN, int RPAD>
__global__ void __launch_bounds__(256)
scan_kernel(const float* __restrict__ pre, const float* __restrict__ Whh,
            float* __restrict__ out, int reverse,
            long so_n, long so_r, long so_t, int chOff) {
  extern __shared__ unsigned char smem[];
  bf16_t* WhhB = (bf16_t*)smem;                                         // 128*512
  bf16_t* hs   = (bf16_t*)(smem + 128 * 512 * 2);                       // 16*128
  float*  gmat = (float*)(smem + 128 * 512 * 2 + 16 * 128 * 2);         // 16*512
  float*  cst  = (float*)(smem + 128 * 512 * 2 + 16 * 128 * 2 + 16 * 512 * 4);

  constexpr int R = 2 * RPN;
  const int r0      = blockIdx.x * 16;
  const int wave    = threadIdx.x >> 5;
  const int nbase   = wave * 64;
  const int lane_hi = (threadIdx.x >> 4) & 1;
  const int lane_lo = threadIdx.x & 15;
  const int h8      = lane_hi * 8;

  // Stage Whh (read coalesced) into fragment-order bf16 LDS.
  for (int e = threadIdx.x; e < 128 * 512; e += 256) {
    const int n = e >> 7, k = e & 127;          // Whh[n][k]
    const int kt = k >> 5, kl = k & 31;
    const int half = kl >> 4, kk = kl & 15;
    WhhB[(((kt * 2 + half) * 32 + (n >> 4)) * 16 + (n & 15)) * 16 + kk] =
        (bf16_t)Whh[e];
  }
  for (int e = threadIdx.x; e < 16 * 128; e += 256) {
    hs[e]  = (bf16_t)0.0f;
    cst[e] = 0.0f;
  }
  __syncthreads();

  // pre C-fragment: v8f at pre[(t*512 + n)*RPAD + r0 + 8*lane_hi]
  auto pre_frag = [&](int t, int nt) -> v8f {
    const int n = nbase + nt * 16 + lane_lo;
    return *(const v8f*)(pre + ((long)t * 512 + n) * RPAD + r0 + h8);
  };

  v8f pc[4];
  {
    const int t0 = reverse ? (T - 1) : 0;
#pragma unroll
    for (int nt = 0; nt < 4; ++nt) pc[nt] = pre_frag(t0, nt);
  }

  for (int tt = 0; tt < T; ++tt) {
    const int t = reverse ? (T - 1 - tt) : tt;
    v8f acc[4];
#pragma unroll
    for (int nt = 0; nt < 4; ++nt) acc[nt] = pc[nt];

    // Prefetch next step's pre while WMMA + gates run.
    const int ttn = (tt + 1 < T) ? (tt + 1) : tt;
    const int tn  = reverse ? (T - 1 - ttn) : ttn;
#pragma unroll
    for (int nt = 0; nt < 4; ++nt) pc[nt] = pre_frag(tn, nt);

#pragma unroll
    for (int kt = 0; kt < 4; ++kt) {
      const v16bf af = load_frag(hs + ((kt * 2 + lane_hi) * 16 + lane_lo) * 16);
#pragma unroll
      for (int nt = 0; nt < 4; ++nt) {
        const v16bf bfg = load_frag(
            WhhB + (((kt * 2 + lane_hi) * 32 + wave * 4 + nt) * 16 + lane_lo) * 16);
        acc[nt] = wmma_bf16(af, bfg, acc[nt]);
      }
    }
#pragma unroll
    for (int nt = 0; nt < 4; ++nt) {
#pragma unroll
      for (int v = 0; v < 8; ++v)
        gmat[(v + h8) * 512 + nbase + nt * 16 + lane_lo] = acc[nt][v];
    }
    __syncthreads();

    for (int e = threadIdx.x; e < 16 * 128; e += 256) {
      const int   m  = e >> 7, j = e & 127;
      const float gi = gmat[m * 512 + j];
      const float gf = gmat[m * 512 + 128 + j];
      const float gg = gmat[m * 512 + 256 + j];
      const float go = gmat[m * 512 + 384 + j];
      const float c  = sigmf(gf) * cst[e] + sigmf(gi) * tanhf(gg);
      cst[e]         = c;
      const float h  = sigmf(go) * tanhf(c);
      hs[swizA(j >> 5, m, j & 31)] = (bf16_t)h;   // A-fragment order for next step
      const int r = r0 + m;
      if (r < R && t >= 1 && t <= T - 2) {
        const int nn = r / RPN, rr = r % RPN;
        out[(long)nn * so_n + (long)rr * so_r + (long)(t - 1) * so_t + chOff + j] = h;
      }
    }
    __syncthreads();
  }
}

// ---------------------------------------------------------------------------
// 1x1 conv -> saliency vector v [2,4800]  (float4 dot over 256 channels)
// ---------------------------------------------------------------------------
__global__ void conv_kernel(const float* __restrict__ x, const float* __restrict__ w,
                            const float* __restrict__ b, float* __restrict__ salv) {
  const int idx = blockIdx.x * blockDim.x + threadIdx.x;
  if (idx >= 2 * 4800) return;
  const float4* p4 = (const float4*)(x + (long)idx * 256);
  const float4* w4 = (const float4*)w;
  float s = b[0];
#pragma unroll 4
  for (int c = 0; c < 64; ++c) {
    const float4 a = p4[c], q = w4[c];
    s += a.x * q.x + a.y * q.y + a.z * q.z + a.w * q.w;
  }
  salv[idx] = s;
}

// ---------------------------------------------------------------------------
// Temporal GEMV: g[n][j] = v[n] . Wih_t[j] + b_t[j].  368MB weight traffic ->
// pure HBM bound; one row per block, float4 coalesced, tree-reduce in LDS.
// ---------------------------------------------------------------------------
__global__ void __launch_bounds__(256)
temporal_gemv_kernel(const float* __restrict__ v, const float* __restrict__ W,
                     const float* __restrict__ bt, float* __restrict__ g) {
  __shared__ float red[2][256];
  const int    j   = blockIdx.x;  // 0..19199
  const float* row = W + (long)j * 4800;
  float s0 = 0.0f, s1 = 0.0f;
  for (int k = (int)threadIdx.x * 4; k < 4800; k += 1024) {
    const float4 w4 = *(const float4*)(row + k);
    const float4 a4 = *(const float4*)(v + k);
    const float4 b4 = *(const float4*)(v + 4800 + k);
    s0 += w4.x * a4.x + w4.y * a4.y + w4.z * a4.z + w4.w * a4.w;
    s1 += w4.x * b4.x + w4.y * b4.y + w4.z * b4.z + w4.w * b4.w;
  }
  red[0][threadIdx.x] = s0;
  red[1][threadIdx.x] = s1;
  __syncthreads();
  for (int off = 128; off > 0; off >>= 1) {
    if (threadIdx.x < off) {
      red[0][threadIdx.x] += red[0][threadIdx.x + off];
      red[1][threadIdx.x] += red[1][threadIdx.x + off];
    }
    __syncthreads();
  }
  if (threadIdx.x == 0) {
    g[j]         = red[0][0] + bt[j];
    g[19200 + j] = red[1][0] + bt[j];
  }
}

// Single-step LSTM from zero state: f-gate vanishes.
__global__ void temporal_gates_kernel(const float* __restrict__ g, float* __restrict__ hmap) {
  const int idx = blockIdx.x * blockDim.x + threadIdx.x;
  if (idx >= 2 * 4800) return;
  const int    n  = idx / 4800, j = idx - n * 4800;
  const float* gb = g + (long)n * 19200;
  const float  gi = gb[j];
  const float  gg = gb[2 * 4800 + j];
  const float  go = gb[3 * 4800 + j];
  const float  c  = sigmf(gi) * tanhf(gg);
  hmap[idx]       = sigmf(go) * tanhf(c);
}

// Bilinear 60x80 -> 480x640, align_corners=True, then sigmoid.
__global__ void upsample_kernel(const float* __restrict__ hmap, float* __restrict__ out) {
  const int idx = blockIdx.x * blockDim.x + threadIdx.x;
  if (idx >= 2 * 480 * 640) return;
  const int n   = idx / (480 * 640);
  const int rem = idx - n * 480 * 640;
  const int oy  = rem / 640, ox = rem - oy * 640;
  const float fy = oy * (59.0f / 479.0f);
  const float fx = ox * (79.0f / 639.0f);
  const int y0 = (int)floorf(fy);
  const int x0 = (int)floorf(fx);
  const int y1 = (y0 + 1 < 59) ? y0 + 1 : 59;
  const int x1 = (x0 + 1 < 79) ? x0 + 1 : 79;
  const float wy = fy - y0, wx = fx - x0;
  const float* hp  = hmap + (long)n * 4800;
  const float  v00 = hp[y0 * 80 + x0], v01 = hp[y0 * 80 + x1];
  const float  v10 = hp[y1 * 80 + x0], v11 = hp[y1 * 80 + x1];
  const float  r0  = v00 * (1.0f - wx) + v01 * wx;
  const float  r1  = v10 * (1.0f - wx) + v11 * wx;
  const float  val = r0 * (1.0f - wy) + r1 * wy;
  out[idx] = 1.0f / (1.0f + __expf(-val));
}

// ---------------------------------------------------------------------------
extern "C" void kernel_launch(void* const* d_in, const int* in_sizes, int n_in,
                              void* d_out, int out_size, void* d_ws, size_t ws_size,
                              hipStream_t stream) {
  const float* local_feats = (const float*)d_in[0];
  const float* scene       = (const float*)d_in[1];
  const float* fc1W        = (const float*)d_in[2];
  const float* fc1b        = (const float*)d_in[3];
  const float* fcRW        = (const float*)d_in[4];
  const float* fcRb        = (const float*)d_in[5];
  const float* h1p[6]; const float* v1p[6]; const float* h2p[6]; const float* v2p[6];
  for (int i = 0; i < 6; ++i) {
    h1p[i] = (const float*)d_in[6 + i];
    v1p[i] = (const float*)d_in[12 + i];
    h2p[i] = (const float*)d_in[18 + i];
    v2p[i] = (const float*)d_in[24 + i];
  }
  const float* convw = (const float*)d_in[30];
  const float* convb = (const float*)d_in[31];
  const float* Wih_t = (const float*)d_in[32];
  const float* b_t   = (const float*)d_in[33];
  float*       out   = (float*)d_out;

  // Workspace carve (~69 MB of floats).
  float* ws   = (float*)d_ws;
  float* ctx1 = ws; ws += 2 * 512;
  float* ctxR = ws; ws += 2 * 256;
  float* xA   = ws; ws += (long)2 * 60 * 80 * 256;
  float* xB   = ws; ws += (long)2 * 60 * 80 * 256;
  float* preF = ws; ws += (long)62 * 512 * 192;   // max of 82*512*128, 62*512*192
  float* preB = ws; ws += (long)62 * 512 * 192;
  float* salv = ws; ws += 2 * 4800;
  float* gbuf = ws; ws += 2 * 19200;
  float* hmap = ws; ws += 2 * 4800;

  ctx_kernel<<<6, 256, 0, stream>>>(scene, fc1W, fc1b, fcRW, fcRb, ctx1, ctxR);

  const size_t scan_lds = 128 * 512 * 2 + 16 * 128 * 2 + 16 * 512 * 4 + 16 * 128 * 4;
  const long   soN = (long)60 * 80 * 256;  // NHWC [2,60,80,256]
  const long   soH = (long)80 * 256;
  const long   soW = 256;

  // ---- pass 1: horizontal, input NCHW [2,512,60,80];  T=82, I=512, RPAD=128
  {
    dim3 g((82 * 128) / 64, 8, 2);
    proj_kernel<82, 512, 60, 128, false><<<g, 128, 0, stream>>>(
        local_feats, ctx1, h1p[0], h1p[3], h1p[2], h1p[5], preF, preB,
        (long)512 * 4800, 80, 1, 4800);
    scan_kernel<82, 60, 128><<<8, 256, scan_lds, stream>>>(
        preF, h1p[1], xA, 0, soN, soH, soW, 0);
    scan_kernel<82, 60, 128><<<8, 256, scan_lds, stream>>>(
        preB, h1p[4], xA, 1, soN, soH, soW, 128);
  }
  // ---- pass 2: vertical on xA;  T=62, I=256, RPAD=192
  {
    dim3 g((62 * 192) / 64, 8, 2);
    proj_kernel<62, 256, 80, 192, true><<<g, 128, 0, stream>>>(
        xA, ctxR, v1p[0], v1p[3], v1p[2], v1p[5], preF, preB,
        soN, soW, soH, 1);
    scan_kernel<62, 80, 192><<<10, 256, scan_lds, stream>>>(
        preF, v1p[1], xB, 0, soN, soW, soH, 0);
    scan_kernel<62, 80, 192><<<10, 256, scan_lds, stream>>>(
        preB, v1p[4], xB, 1, soN, soW, soH, 128);
  }
  // ---- pass 3: horizontal on xB;  T=82, I=256, RPAD=128
  {
    dim3 g((82 * 128) / 64, 8, 2);
    proj_kernel<82, 256, 60, 128, true><<<g, 128, 0, stream>>>(
        xB, ctxR, h2p[0], h2p[3], h2p[2], h2p[5], preF, preB,
        soN, soH, soW, 1);
    scan_kernel<82, 60, 128><<<8, 256, scan_lds, stream>>>(
        preF, h2p[1], xA, 0, soN, soH, soW, 0);
    scan_kernel<82, 60, 128><<<8, 256, scan_lds, stream>>>(
        preB, h2p[4], xA, 1, soN, soH, soW, 128);
  }
  // ---- pass 4: vertical on xA;  T=62, I=256, RPAD=192
  {
    dim3 g((62 * 192) / 64, 8, 2);
    proj_kernel<62, 256, 80, 192, true><<<g, 128, 0, stream>>>(
        xA, ctxR, v2p[0], v2p[3], v2p[2], v2p[5], preF, preB,
        soN, soW, soH, 1);
    scan_kernel<62, 80, 192><<<10, 256, scan_lds, stream>>>(
        preF, v2p[1], xB, 0, soN, soW, soH, 0);
    scan_kernel<62, 80, 192><<<10, 256, scan_lds, stream>>>(
        preB, v2p[4], xB, 1, soN, soW, soH, 128);
  }

  conv_kernel<<<(9600 + 255) / 256, 256, 0, stream>>>(xB, convw, convb, salv);
  temporal_gemv_kernel<<<19200, 256, 0, stream>>>(salv, Wih_t, b_t, gbuf);
  temporal_gates_kernel<<<(9600 + 255) / 256, 256, 0, stream>>>(gbuf, hmap);
  upsample_kernel<<<(2 * 480 * 640 + 255) / 256, 256, 0, stream>>>(hmap, out);
}
